// BruteForceLayer_15736760172796
// MI455X (gfx1250) — matrix operand order repaired and checked by reading, not compiled, not run on hardware
//
#include <hip/hip_runtime.h>

// ---------------------------------------------------------------------------
// Brute-force top-k: scores = Q[512,64] . C^T[64,1e6], top-100 per query.
// GEMM via V_WMMA_F32_16X16X32_BF16 with hi/lo bf16 split of f32 inputs.
// Selection via 2-pass radix-select (histogram -> threshold -> survivors).
// ---------------------------------------------------------------------------

typedef __attribute__((ext_vector_type(16))) __bf16 v16bf;
typedef __attribute__((ext_vector_type(8)))  float  v8f;

namespace {
constexpr int QB    = 512;            // #queries
constexpr int DD    = 64;             // dim
constexpr int NN    = 1000000;        // #candidates
constexpr int TOPK  = 100;
constexpr int TILES = NN / 16;        // 62500 candidate tiles of 16
constexpr int BINS  = 4096;           // 12-bit prefix bins of ordered f32 bits
constexpr int CAP   = 4096;           // survivor capacity per query
constexpr int WAVES = 16;             // waves per score block (wave32)
constexpr int TPB   = WAVES * 32;     // 512 threads
constexpr int GRIDX1 = 64;            // pass-1 grid (LDS-occupancy limited)
constexpr int GRIDX2 = 128;           // pass-2 grid (no LDS -> more waves)
}

__device__ __forceinline__ unsigned ordf(float f) {
  // monotone map f32 -> u32 (greater float -> greater uint)
  unsigned u = __float_as_uint(f);
  return (u & 0x80000000u) ? ~u : (u | 0x80000000u);
}

__device__ __forceinline__ void split_bf16(float x, __bf16 &hi, __bf16 &lo) {
  hi = (__bf16)x;
  lo = (__bf16)(x - (float)hi);
}

// ---------------------------------------------------------------------------
// Pre-pass: pack candidates into WMMA B-operand-ready hi/lo bf16 layout.
// Per (tile t, k-half h, lane): 16 hi bf16 then 16 lo bf16 (64 B contiguous).
// Code's B k order (contiguous, same as A, not ISA table): lanes 0-15 hold K=0..15 of column `lane`,
// lanes 16-31 hold K=16..31 of column `lane-16`.
// ---------------------------------------------------------------------------
__global__ __launch_bounds__(256) void preconv_kernel(
    const float* __restrict__ cand, __bf16* __restrict__ pre) {
  size_t g = (size_t)blockIdx.x * blockDim.x + threadIdx.x;  // over TILES*2*32
  if (g >= (size_t)TILES * 64) return;
  int lane = (int)(g & 31);
  int h    = (int)((g >> 5) & 1);
  int t    = (int)(g >> 6);
  int wl = lane & 15, sel = lane >> 4;
  const float4* c4 =
      (const float4*)(cand + (size_t)(t * 16 + wl) * DD + h * 32 + sel * 16);
  v16bf bhi, blo;
#pragma unroll
  for (int v = 0; v < 4; ++v) {
    float4 xv = c4[v];
    float xs[4] = {xv.x, xv.y, xv.z, xv.w};
#pragma unroll
    for (int e = 0; e < 4; ++e) {
      __bf16 hi, lo;
      split_bf16(xs[e], hi, lo);
      bhi[v * 4 + e] = hi;
      blo[v * 4 + e] = lo;
    }
  }
  __bf16* p = pre + g * 32;          // 64B-aligned
  *(v16bf*)p = bhi;                  // 32B vector store
  *(v16bf*)(p + 16) = blo;           // 32B vector store
}

// ---------------------------------------------------------------------------
// Score kernel (shared by both passes).
//   PASS==1: build per-query histogram of score bit-prefix (LDS -> global).
//   PASS==2: append (score, idx) survivors >= per-query threshold.
// Each wave owns whole 16x16 output tiles -> EXEC all-ones around WMMA.
// Score bits are identical across passes: per-tile accumulation order is
// fixed by the code, independent of grid shape.
// ---------------------------------------------------------------------------
template <bool PRECONV, int PASS, int GX>
__global__ __launch_bounds__(TPB) void score_kernel(
    const float* __restrict__ q, const float* __restrict__ cand,
    const __bf16* __restrict__ pre,
    unsigned* __restrict__ ghist,            // pass 1
    const unsigned* __restrict__ thrU,       // pass 2
    unsigned* __restrict__ cnt, float* __restrict__ sval,
    int* __restrict__ sidx) {
  extern __shared__ unsigned shist[];        // 16 * BINS (pass 1 only)

  const int tid  = threadIdx.x;
  const int w    = tid >> 5;
  const int lane = tid & 31;
  const int wl   = lane & 15;
  const int sel  = lane >> 4;
  const int q0   = blockIdx.y * 16;

  if constexpr (PASS == 1) {
    for (int i = tid; i < 16 * BINS; i += TPB) shist[i] = 0u;
    __syncthreads();
  }

  // A operand (queries), split hi/lo, held for the whole chunk.
  // Code's A k order (contiguous, same as B, not ISA table): lanes 0-15 row M=lane hold K=0..15,
  // lanes 16-31 row M=lane-16 hold K=16..31 (packed 2/VGPR).
  v16bf qhi[2], qlo[2];
  {
    const float* qrow = q + (size_t)(q0 + wl) * DD;
#pragma unroll
    for (int h = 0; h < 2; ++h) {
      const int b1 = h * 32 + sel * 16;
#pragma unroll
      for (int j = 0; j < 8; ++j) {
        __bf16 h1, l1, h2, l2;
        split_bf16(qrow[b1 + j], h1, l1);
        split_bf16(qrow[b1 + 8 + j], h2, l2);
        qhi[h][j] = h1;     qlo[h][j] = l1;
        qhi[h][8 + j] = h2; qlo[h][8 + j] = l2;
      }
    }
  }

  // Pass-2: hoist the 8 per-lane thresholds out of the tile loop.
  unsigned thr[8];
  if constexpr (PASS == 2) {
#pragma unroll
    for (int r = 0; r < 8; ++r) thr[r] = thrU[q0 + sel * 8 + r];
  }

  constexpr int TSTRIDE = GX * WAVES;
  for (int t = blockIdx.x * WAVES + w; t < TILES; t += TSTRIDE) {
    if constexpr (PRECONV) {
      // Prefetch next tile this wave will touch (streaming, > L2 capacity).
      const int tn = t + TSTRIDE;
      if (tn < TILES) {
        const __bf16* pn = pre + ((size_t)(tn * 2) * 32 + lane) * 32;
        __builtin_prefetch(pn, 0, 1);
        __builtin_prefetch(pn + 1024, 0, 1);  // second k-half block (+2048B)
      }
    }
    v8f acc = {};
#pragma unroll
    for (int h = 0; h < 2; ++h) {
      v16bf bhi, blo;
      if constexpr (PRECONV) {
        const __bf16* p = pre + ((size_t)(t * 2 + h) * 32 + lane) * 32;
        bhi = *(const v16bf*)p;          // 32B, 64B-aligned
        blo = *(const v16bf*)(p + 16);   // 32B-aligned
      } else {
        const float4* c4 = (const float4*)(cand + (size_t)(t * 16 + wl) * DD +
                                           h * 32 + sel * 16);
#pragma unroll
        for (int v = 0; v < 4; ++v) {
          float4 xv = c4[v];
          float xs[4] = {xv.x, xv.y, xv.z, xv.w};
#pragma unroll
          for (int e = 0; e < 4; ++e) {
            __bf16 hi, lo;
            split_bf16(xs[e], hi, lo);
            bhi[v * 4 + e] = hi;
            blo[v * 4 + e] = lo;
          }
        }
      }
      // score ~= qhi*bhi + qhi*blo + qlo*bhi   (drop lo*lo)
      acc = __builtin_amdgcn_wmma_f32_16x16x32_bf16(
          false, qhi[h], false, bhi, (short)0, acc, false, false);
      acc = __builtin_amdgcn_wmma_f32_16x16x32_bf16(
          false, qhi[h], false, blo, (short)0, acc, false, false);
      acc = __builtin_amdgcn_wmma_f32_16x16x32_bf16(
          false, qlo[h], false, bhi, (short)0, acc, false, false);
    }

    // C/D layout: lane holds column N=wl (candidate t*16+wl),
    // VGPR r -> row M = r + sel*8 (query q0 + r + sel*8).
    const int ci = t * 16 + wl;
#pragma unroll
    for (int r = 0; r < 8; ++r) {
      const float s = acc[r];
      const unsigned u = ordf(s);
      if constexpr (PASS == 1) {
        atomicAdd(&shist[(r + sel * 8) * BINS + (int)(u >> 20)], 1u);
      } else {
        if (u >= thr[r]) {
          const int qq = q0 + sel * 8 + r;
          unsigned pos = atomicAdd(&cnt[qq], 1u);
          if (pos < (unsigned)CAP) {
            sval[(size_t)qq * CAP + pos] = s;
            sidx[(size_t)qq * CAP + pos] = ci;
          }
        }
      }
    }
  }

  if constexpr (PASS == 1) {
    __syncthreads();
    for (int i = tid; i < 16 * BINS; i += TPB) {
      unsigned c = shist[i];
      if (c)
        atomicAdd(&ghist[(size_t)(q0 + i / BINS) * BINS + (i % BINS)], c);
    }
  }
}

// ---------------------------------------------------------------------------
// Per-query threshold: largest bin b with count(>= b) >= TOPK.
// ---------------------------------------------------------------------------
__global__ __launch_bounds__(256) void thresh_kernel(
    const unsigned* __restrict__ ghist, unsigned* __restrict__ thrU) {
  int qq = blockIdx.x * blockDim.x + threadIdx.x;
  if (qq >= QB) return;
  const unsigned* hg = ghist + (size_t)qq * BINS;
  unsigned cum = 0;
  int b = BINS - 1;
  for (; b > 0; --b) {
    cum += hg[b];
    if (cum >= (unsigned)TOPK) break;
  }
  thrU[qq] = (unsigned)b << 20;   // lower edge of bin b in ordered-uint space
}

// ---------------------------------------------------------------------------
// Exact top-100 over survivors: iterative block argmax (descending order,
// ties broken by smaller index, matching jax.lax.top_k).
// ---------------------------------------------------------------------------
__global__ __launch_bounds__(256) void topk_kernel(
    const unsigned* __restrict__ cnt, const float* __restrict__ sval,
    const int* __restrict__ sidx, float* __restrict__ outv,
    int* __restrict__ outi) {
  __shared__ float v[CAP];
  __shared__ int   ix[CAP];
  __shared__ float rv[256];
  __shared__ int   ri[256];
  __shared__ int   rp[256];

  const int qq = blockIdx.x;
  const int tid = threadIdx.x;
  const float NEG = __int_as_float(0xff800000);  // -inf

  int n = (int)cnt[qq];
  if (n > CAP) n = CAP;
  for (int i = tid; i < n; i += 256) {
    v[i]  = sval[(size_t)qq * CAP + i];
    ix[i] = sidx[(size_t)qq * CAP + i];
  }
  __syncthreads();

  for (int j = 0; j < TOPK; ++j) {
    float bv = NEG;
    int bi = 0x7fffffff, bp = -1;
    for (int i = tid; i < n; i += 256) {
      float cv = v[i];
      int cix = ix[i];
      if (cv > bv || (cv == bv && cix < bi)) { bv = cv; bi = cix; bp = i; }
    }
    rv[tid] = bv; ri[tid] = bi; rp[tid] = bp;
    __syncthreads();
    for (int s = 128; s > 0; s >>= 1) {
      if (tid < s) {
        if (rv[tid + s] > rv[tid] ||
            (rv[tid + s] == rv[tid] && ri[tid + s] < ri[tid])) {
          rv[tid] = rv[tid + s]; ri[tid] = ri[tid + s]; rp[tid] = rp[tid + s];
        }
      }
      __syncthreads();
    }
    if (tid == 0) {
      outv[(size_t)qq * TOPK + j] = rv[0];
      outi[(size_t)qq * TOPK + j] = ri[0];
      if (rp[0] >= 0) { v[rp[0]] = NEG; ix[rp[0]] = 0x7fffffff; }
    }
    __syncthreads();
  }
}

// ---------------------------------------------------------------------------
extern "C" void kernel_launch(void* const* d_in, const int* in_sizes, int n_in,
                              void* d_out, int out_size, void* d_ws,
                              size_t ws_size, hipStream_t stream) {
  (void)in_sizes; (void)n_in; (void)out_size;
  const float* qv   = (const float*)d_in[0];
  const float* cand = (const float*)d_in[1];
  // d_in[2] is k==100 on device; TOPK is fixed at compile time per reference.

  float* outv = (float*)d_out;
  int*   outi = (int*)d_out + (size_t)QB * TOPK;

  char* ws = (char*)d_ws;
  size_t off = 0;
  auto take = [&](size_t bytes) {
    void* p = ws + off;
    off = (off + bytes + 255) & ~(size_t)255;
    return p;
  };
  unsigned* ghist = (unsigned*)take((size_t)QB * BINS * sizeof(unsigned));
  unsigned* thrU  = (unsigned*)take((size_t)QB * sizeof(unsigned));
  unsigned* cnt   = (unsigned*)take((size_t)QB * sizeof(unsigned));
  size_t zero_end = off;                       // zero hist + thr + cnt
  float* sval = (float*)take((size_t)QB * CAP * sizeof(float));
  int*   sidx = (int*)take((size_t)QB * CAP * sizeof(int));
  __bf16* pre = (__bf16*)(ws + off);
  size_t need_big = off + (size_t)TILES * 2 * 32 * 32 * sizeof(__bf16);  // 256MB
  const bool use_pre = (ws_size >= need_big);

  hipMemsetAsync(d_ws, 0, zero_end, stream);

  if (use_pre) {
    int total = TILES * 64;
    preconv_kernel<<<dim3((total + 255) / 256), dim3(256), 0, stream>>>(cand,
                                                                        pre);
  }

  const size_t histBytes = (size_t)16 * BINS * sizeof(unsigned);  // 256 KB LDS
  const dim3 sg1(GRIDX1, QB / 16);
  const dim3 sg2(GRIDX2, QB / 16);
  if (use_pre) {
    score_kernel<true, 1, GRIDX1><<<sg1, TPB, histBytes, stream>>>(
        qv, cand, pre, ghist, nullptr, nullptr, nullptr, nullptr);
  } else {
    score_kernel<false, 1, GRIDX1><<<sg1, TPB, histBytes, stream>>>(
        qv, cand, nullptr, ghist, nullptr, nullptr, nullptr, nullptr);
  }

  thresh_kernel<<<dim3((QB + 255) / 256), dim3(256), 0, stream>>>(ghist, thrU);

  if (use_pre) {
    score_kernel<true, 2, GRIDX2><<<sg2, TPB, 0, stream>>>(
        qv, cand, pre, nullptr, thrU, cnt, sval, sidx);
  } else {
    score_kernel<false, 2, GRIDX2><<<sg2, TPB, 0, stream>>>(
        qv, cand, nullptr, nullptr, thrU, cnt, sval, sidx);
  }

  topk_kernel<<<dim3(QB), dim3(256), 0, stream>>>(cnt, sval, sidx, outv, outi);
}
